// GeoS4FormerNetV2_34402688041063
// MI455X (gfx1250) — matrix operand order repaired and checked by reading, not compiled, not run on hardware
//
#include <hip/hip_runtime.h>
#include <hip/hip_bf16.h>
#include <math.h>

namespace cfg {
constexpr int BATCH = 128;
constexpr int SEQL  = 2048;
constexpr int HID   = 64;
constexpr int NHEAD = 2;
constexpr int DHEAD = 32;
constexpr int NNODE = 50000;
constexpr int NEDGE = 800000;
constexpr int ETOT  = NEDGE + NNODE;   // with self loops
constexpr int NST   = 64;
constexpr int NLAY  = 3;
}

typedef __attribute__((ext_vector_type(16))) __bf16 v16bf;
typedef __attribute__((ext_vector_type(8)))  float  v8f;

union BF16x16 { v16bf v; unsigned short u[16]; };

__device__ inline unsigned short f2bf(float f) {
  unsigned int x = __float_as_uint(f);
  unsigned int r = x + 0x7FFFu + ((x >> 16) & 1u);   // round-to-nearest-even
  return (unsigned short)(r >> 16);
}

__device__ inline float gelu_tanh(float x) {
  float x3 = x * x * x;
  return 0.5f * x * (1.0f + tanhf(0.7978845608028654f * (x + 0.044715f * x3)));
}
__device__ inline float nan2num(float x) {
  if (isnan(x)) return 0.0f;
  if (isinf(x)) return x > 0.0f ? 1e6f : -1e6f;
  return x;
}
__device__ inline float wave_sum32(float v) {
  #pragma unroll
  for (int m = 16; m > 0; m >>= 1) v += __shfl_xor(v, m, 32);
  return v;
}
__device__ inline float block_sum64(float v, volatile float* buf) {
  int t = threadIdx.x;
  buf[t] = v; __syncthreads();
  if (t < 32) buf[t] += buf[t + 32]; __syncthreads();
  if (t < 16) buf[t] += buf[t + 16]; __syncthreads();
  if (t <  8) buf[t] += buf[t +  8]; __syncthreads();
  if (t <  4) buf[t] += buf[t +  4]; __syncthreads();
  if (t <  2) buf[t] += buf[t +  2]; __syncthreads();
  if (t <  1) buf[t] += buf[t +  1]; __syncthreads();
  float r = buf[0]; __syncthreads();
  return r;
}
__device__ inline void atomicMaxF(float* addr, float val) {
  int* ia = (int*)addr;
  int old = __float_as_int(*addr);
  while (__int_as_float(old) < val) {
    int assumed = old;
    old = atomicCAS(ia, assumed, __float_as_int(val));
    if (old == assumed) break;
  }
}
__device__ inline void edge_sd(const int* __restrict__ ei, int e, int& s, int& d) {
  if (e < cfg::NEDGE) { s = ei[e]; d = ei[cfg::NEDGE + e]; }
  else { s = e - cfg::NEDGE; d = s; }
}

// ---------------------------------------------------------------- temporal ---

// Per (h,n): discrete pole a = exp(dt*A), effective C, geometric bias sum G.
__global__ void s4_precompute(const float* __restrict__ log_dt,
                              const float* __restrict__ A_re, const float* __restrict__ A_im,
                              const float* __restrict__ C_re, const float* __restrict__ C_im,
                              float* a_re, float* a_im, float* cf_re, float* cf_im,
                              float* g_re, float* g_im) {
  int idx = blockIdx.x * blockDim.x + threadIdx.x;
  if (idx >= cfg::HID * cfg::NST) return;
  int h = idx >> 6;
  float dt  = expf(log_dt[h]);
  float are = A_re[idx], aim = A_im[idx];
  float dr = dt * are, di = dt * aim;
  float er = expf(dr);
  float ar = er * cosf(di), ai = er * sinf(di);          // a = exp(dtA)
  // Ceff = C * (exp(dtA)-1)/A
  float Er = ar - 1.0f, Ei = ai;
  float den = are * are + aim * aim;
  float qr = (Er * are + Ei * aim) / den;
  float qi = (Ei * are - Er * aim) / den;
  float cr = C_re[idx], ci = C_im[idx];
  // G = (1 - a^L) / (1 - a)
  float Lr = (float)cfg::SEQL * dr, Li = (float)cfg::SEQL * di;
  float eL = expf(Lr);
  float nr = 1.0f - eL * cosf(Li), ni = -eL * sinf(Li);
  float dr2 = 1.0f - ar, di2 = -ai;
  float dd = dr2 * dr2 + di2 * di2 + 1e-30f;
  a_re[idx]  = ar;  a_im[idx]  = ai;
  cf_re[idx] = cr * qr - ci * qi;
  cf_im[idx] = cr * qi + ci * qr;
  g_re[idx]  = (nr * dr2 + ni * di2) / dd;
  g_im[idx]  = (ni * dr2 - nr * di2) / dd;
}

// One block (64 threads) per (b,h); one complex state per thread; scan over L.
__global__ void s4_scan(const float* __restrict__ x,
                        const float* __restrict__ a_re, const float* __restrict__ a_im,
                        const float* __restrict__ cf_re, const float* __restrict__ cf_im,
                        const float* __restrict__ g_re, const float* __restrict__ g_im,
                        const float* __restrict__ enc_w, const float* __restrict__ enc_b,
                        const float* __restrict__ Dp, float* __restrict__ y_last) {
  __shared__ float xs[512];
  __shared__ float red[64];
  int b = blockIdx.x, h = blockIdx.y;
  int idx = h * cfg::NST + threadIdx.x;
  const float* xb = x + (long long)b * cfg::SEQL;
  float ar = a_re[idx], ai = a_im[idx];
  float sr = 0.0f, si = 0.0f;
  for (int c0 = 0; c0 < cfg::SEQL; c0 += 512) {
    for (int i = threadIdx.x; i < 512; i += 64) xs[i] = xb[c0 + i];
    if (c0 + 512 < cfg::SEQL) __builtin_prefetch(xb + c0 + 512, 0, 0);
    __syncthreads();
    #pragma unroll 4
    for (int i = 0; i < 512; ++i) {
      float xv = xs[i];
      float nr = fmaf(ar, sr, fmaf(-ai, si, xv));
      float ni = fmaf(ar, si, ai * sr);
      sr = nr; si = ni;
    }
    __syncthreads();
  }
  float w = enc_w[h], be = enc_b[h];
  float su_r = fmaf(w, sr, be * g_re[idx]);
  float su_i = fmaf(w, si, be * g_im[idx]);
  float contrib = cf_re[idx] * su_r - cf_im[idx] * su_i;
  float tot = block_sum64(contrib, red);
  if (threadIdx.x == 0) {
    float u_last = fmaf(xb[cfg::SEQL - 1], w, be);
    y_last[b * cfg::HID + h] = 2.0f * tot + u_last * Dp[h];
  }
}

// Remainder of the temporal branch at t = L-1 (mix, LN, dec, TCN, LN).
__global__ void s4_tail(const float* __restrict__ x, const float* __restrict__ y_last,
                        const float* __restrict__ enc_w, const float* __restrict__ enc_b,
                        const float* __restrict__ mix_w, const float* __restrict__ mix_b,
                        const float* __restrict__ ln_g, const float* __restrict__ ln_b,
                        const float* __restrict__ dec_w, const float* __restrict__ dec_b,
                        const float* __restrict__ tcn_w, const float* __restrict__ tcn_b,
                        const float* __restrict__ tln_g, const float* __restrict__ tln_b,
                        float* __restrict__ ht) {
  __shared__ float gy[64], xsv[64], red[64];
  int b = blockIdx.x, j = threadIdx.x;
  gy[j] = gelu_tanh(y_last[b * 64 + j]);
  __syncthreads();
  float z = mix_b[j];
  #pragma unroll 8
  for (int d = 0; d < 64; ++d) z = fmaf(gy[d], mix_w[d * 64 + j], z);
  const float* xb = x + (long long)b * cfg::SEQL;
  float xm3 = xb[cfg::SEQL - 3], xm1 = xb[cfg::SEQL - 1];
  float u_last = fmaf(xm1, enc_w[j], enc_b[j]);
  float v = u_last + z;
  float m  = block_sum64(v, red) * (1.0f / 64.0f);
  float vr = block_sum64((v - m) * (v - m), red) * (1.0f / 64.0f);
  float xsj = (v - m) * rsqrtf(vr + 1e-5f) * ln_g[j] + ln_b[j];
  xsv[j] = xsj;
  __syncthreads();
  float s4o = dec_b[j];
  #pragma unroll 8
  for (int d = 0; d < 64; ++d) s4o = fmaf(xsv[d], dec_w[d * 64 + j], s4o);
  float tcn = fmaf(tcn_w[j * 3 + 0], xm3, fmaf(tcn_w[j * 3 + 1], xm1, tcn_b[j]));
  float t = nan2num(s4o + tcn);
  float m2  = block_sum64(t, red) * (1.0f / 64.0f);
  float vr2 = block_sum64((t - m2) * (t - m2), red) * (1.0f / 64.0f);
  ht[b * 64 + j] = (t - m2) * rsqrtf(vr2 + 1e-5f) * tln_g[j] + tln_b[j];
}

// ----------------------------------------------------------------- spatial ---

__global__ void gat_init_xg(const float* __restrict__ coords,
                            const float* __restrict__ gw, const float* __restrict__ gb,
                            float* __restrict__ xg) {
  long long idx = (long long)blockIdx.x * blockDim.x + threadIdx.x;
  if (idx >= (long long)cfg::NNODE * 64) return;
  int n = (int)(idx >> 6), f = (int)(idx & 63);
  xg[idx] = fmaf(coords[n * 2 + 0], gw[f], fmaf(coords[n * 2 + 1], gw[64 + f], gb[f]));
}

__global__ void gat_buf_init(float* __restrict__ agg, float* __restrict__ emax,
                             float* __restrict__ denom) {
  long long idx = (long long)blockIdx.x * blockDim.x + threadIdx.x;
  if (idx >= (long long)cfg::NNODE * 64) return;
  agg[idx] = 0.0f;
  if (idx < (long long)cfg::NNODE * 2) { emax[idx] = -1e30f; denom[idx] = 0.0f; }
}

// xl = xg @ wl, xr = xg @ wr via V_WMMA_F32_16X16X32_BF16.
// One wave per 16-row tile: the A fragments (two K-steps of 16x32) are loaded
// into registers ONCE and reused across both weight matrices and all four
// 16-column output tiles -> 16 back-to-back WMMAs per wave, 8x less xg traffic.
__global__ void gat_proj_wmma(const float* __restrict__ xg,
                              const float* __restrict__ wl, const float* __restrict__ wr,
                              float* __restrict__ xl, float* __restrict__ xr) {
  int tile = blockIdx.x;        // NNODE/16 = 3125 tiles exactly
  int lane = threadIdx.x;       // single wave32
  int m  = lane & 15;
  int hi = lane >> 4;
  int row = tile * 16 + m;
  // A 16x32 bf16 fragments: lane row m; K = kb + hi*8 + e (+8 for e>=8)
  BF16x16 a0, a1;
  #pragma unroll
  for (int e = 0; e < 16; ++e) {
    int ka = hi * 8 + e + ((e >= 8) ? 8 : 0);
    a0.u[e] = f2bf(xg[row * 64 + ka]);
    a1.u[e] = f2bf(xg[row * 64 + 32 + ka]);
  }
  #pragma unroll
  for (int mat = 0; mat < 2; ++mat) {
    const float* Wm = mat ? wr : wl;
    float* O = mat ? xr : xl;
    #pragma unroll
    for (int ct = 0; ct < 4; ++ct) {
      v8f c = {};
      #pragma unroll
      for (int ks = 0; ks < 2; ++ks) {
        // B 32x16 bf16 fragment: lane col m; K = ks*32 + hi*8 + e (+8 for e>=8)
        BF16x16 bfr;
        #pragma unroll
        for (int e = 0; e < 16; ++e) {
          int kw = ks * 32 + hi * 8 + e + ((e >= 8) ? 8 : 0);
          bfr.u[e] = f2bf(Wm[kw * 64 + ct * 16 + m]);
        }
#ifdef __AMDGCN__
        c = __builtin_amdgcn_wmma_f32_16x16x32_bf16(false, ks ? a1.v : a0.v,
                                                    false, bfr.v,
                                                    (short)0, c, false, false);
#endif
      }
      #pragma unroll
      for (int r = 0; r < 8; ++r) {
        int mm = r + 8 * hi;    // D: vgpr r -> row r (+8 for upper lanes)
        O[(tile * 16 + mm) * 64 + ct * 16 + m] = c[r];
      }
    }
  }
}

// Pass 1: e[edge,h] = att_h . leakyrelu(xl[src]+xr[dst]); segment max over dst.
__global__ void gat_edge_score(const int* __restrict__ ei,
                               const float* __restrict__ xl, const float* __restrict__ xr,
                               const float* __restrict__ att,
                               float* __restrict__ esc, float* __restrict__ emax) {
  int wid = threadIdx.x >> 5, lane = threadIdx.x & 31;
  int e = blockIdx.x * 8 + wid;
  if (e >= cfg::ETOT) return;
  int s, d; edge_sd(ei, e, s, d);
  #pragma unroll
  for (int h = 0; h < 2; ++h) {
    float v = xl[s * 64 + h * 32 + lane] + xr[d * 64 + h * 32 + lane];
    v = v > 0.0f ? v : 0.2f * v;
    float sum = wave_sum32(v * att[h * 32 + lane]);
    if (lane == 0) {
      esc[e * 2 + h] = sum;
      atomicMaxF(&emax[d * 2 + h], sum);
    }
  }
}

// Pass 2: ex = exp(e - emax[dst]); segment sum into denom.
__global__ void gat_edge_soft(const int* __restrict__ ei,
                              float* __restrict__ esc, const float* __restrict__ emax,
                              float* __restrict__ denom) {
  int idx = blockIdx.x * blockDim.x + threadIdx.x;
  if (idx >= cfg::ETOT * 2) return;
  int e = idx >> 1, h = idx & 1;
  int s, d; edge_sd(ei, e, s, d); (void)s;
  float v = expf(esc[idx] - emax[d * 2 + h]);
  esc[idx] = v;
  atomicAdd(&denom[d * 2 + h], v);
}

// Pass 3: out[dst] += alpha * xl[src]
__global__ void gat_edge_aggr(const int* __restrict__ ei,
                              const float* __restrict__ xl, const float* __restrict__ esc,
                              const float* __restrict__ denom, float* __restrict__ agg) {
  int wid = threadIdx.x >> 5, lane = threadIdx.x & 31;
  int e = blockIdx.x * 8 + wid;
  if (e >= cfg::ETOT) return;
  int s, d; edge_sd(ei, e, s, d);
  #pragma unroll
  for (int h = 0; h < 2; ++h) {
    float alpha = esc[e * 2 + h] / (denom[d * 2 + h] + 1e-16f);
    atomicAdd(&agg[d * 64 + h * 32 + lane], alpha * xl[s * 64 + h * 32 + lane]);
  }
}

// Pass 4: xg = nan2num(LN(elu(agg + bias) + xg))  (wave per node, 2 feats/lane)
__global__ void gat_node_update(const float* __restrict__ agg, const float* __restrict__ bias,
                                const float* __restrict__ lg, const float* __restrict__ lb,
                                float* __restrict__ xg) {
  int wid = threadIdx.x >> 5, lane = threadIdx.x & 31;
  int n = blockIdx.x * 8 + wid;
  if (n >= cfg::NNODE) return;
  int f0 = lane, f1 = lane + 32;
  float t0 = agg[n * 64 + f0] + bias[f0];
  float t1 = agg[n * 64 + f1] + bias[f1];
  t0 = t0 > 0.0f ? t0 : expf(t0) - 1.0f;
  t1 = t1 > 0.0f ? t1 : expf(t1) - 1.0f;
  float v0 = t0 + xg[n * 64 + f0];
  float v1 = t1 + xg[n * 64 + f1];
  float m  = wave_sum32(v0 + v1) * (1.0f / 64.0f);
  float vr = wave_sum32((v0 - m) * (v0 - m) + (v1 - m) * (v1 - m)) * (1.0f / 64.0f);
  float rs = rsqrtf(vr + 1e-5f);
  xg[n * 64 + f0] = nan2num((v0 - m) * rs * lg[f0] + lb[f0]);
  xg[n * 64 + f1] = nan2num((v1 - m) * rs * lg[f1] + lb[f1]);
}

__global__ void gat_gather(const float* __restrict__ xg, const int* __restrict__ nidx,
                           float* __restrict__ hg) {
  int idx = blockIdx.x * blockDim.x + threadIdx.x;
  if (idx >= cfg::BATCH * 64) return;
  int b = idx >> 6, f = idx & 63;
  hg[idx] = xg[(long long)nidx[b] * 64 + f];
}

// ------------------------------------------------------------------ extras ---

__global__ void extras_mlp(const float* __restrict__ ex,
                           const float* __restrict__ w1, const float* __restrict__ b1,
                           const float* __restrict__ w2, const float* __restrict__ b2,
                           float* __restrict__ he) {
  __shared__ float e[16], h1[64];
  int b = blockIdx.x, j = threadIdx.x;
  if (j < 16) e[j] = ex[b * 16 + j];
  __syncthreads();
  float s = b1[j];
  #pragma unroll
  for (int d = 0; d < 16; ++d) s = fmaf(e[d], w1[d * 64 + j], s);
  h1[j] = nan2num(fmaxf(s, 0.0f));
  __syncthreads();
  float o = b2[j];
  #pragma unroll 8
  for (int d = 0; d < 64; ++d) o = fmaf(h1[d], w2[d * 64 + j], o);
  he[b * 64 + j] = o;
}

// ------------------------------------------------------------------- mixer ---

__global__ void mixer_head(const float* __restrict__ ht, const float* __restrict__ hg,
                           const float* __restrict__ he,
                           const float* __restrict__ wq, const float* __restrict__ bq,
                           const float* __restrict__ wk, const float* __restrict__ bk,
                           const float* __restrict__ wv, const float* __restrict__ bv,
                           const float* __restrict__ wo, const float* __restrict__ bo,
                           const float* __restrict__ l1g, const float* __restrict__ l1b,
                           const float* __restrict__ fw1, const float* __restrict__ fb1,
                           const float* __restrict__ fw2, const float* __restrict__ fb2,
                           const float* __restrict__ l2g, const float* __restrict__ l2b,
                           const float* __restrict__ hw1, const float* __restrict__ hb1,
                           const float* __restrict__ hw2, const float* __restrict__ hb2,
                           float* __restrict__ out) {
  __shared__ float tok[3][64], qm[3][64], km[3][64], vm[3][64], om[3][64], xm[3][64];
  __shared__ float f1[3][128], hf[64], h1s[32], red[64], alp[3][2][3];
  int b = blockIdx.x, j = threadIdx.x;
  tok[0][j] = ht[b * 64 + j]; tok[1][j] = hg[b * 64 + j]; tok[2][j] = he[b * 64 + j];
  __syncthreads();
  for (int t = 0; t < 3; ++t) {
    float q = bq[j], k = bk[j], v = bv[j];
    #pragma unroll 8
    for (int d = 0; d < 64; ++d) {
      float x = tok[t][d];
      q = fmaf(x, wq[d * 64 + j], q);
      k = fmaf(x, wk[d * 64 + j], k);
      v = fmaf(x, wv[d * 64 + j], v);
    }
    qm[t][j] = q; km[t][j] = k; vm[t][j] = v;
  }
  __syncthreads();
  if (j < 6) {                                   // scores + softmax per (t, head)
    int t = j >> 1, h = j & 1;
    float sc[3];
    for (int tp = 0; tp < 3; ++tp) {
      float s = 0.0f;
      #pragma unroll 8
      for (int d = 0; d < 32; ++d) s = fmaf(qm[t][h * 32 + d], km[tp][h * 32 + d], s);
      sc[tp] = s * 0.17677669529663687f;         // 1/sqrt(32)
    }
    float mx = fmaxf(sc[0], fmaxf(sc[1], sc[2]));
    float e0 = expf(sc[0] - mx), e1 = expf(sc[1] - mx), e2 = expf(sc[2] - mx);
    float inv = 1.0f / (e0 + e1 + e2);
    alp[t][h][0] = e0 * inv; alp[t][h][1] = e1 * inv; alp[t][h][2] = e2 * inv;
  }
  __syncthreads();
  int hh = j >> 5;
  for (int t = 0; t < 3; ++t)
    om[t][j] = alp[t][hh][0] * vm[0][j] + alp[t][hh][1] * vm[1][j] + alp[t][hh][2] * vm[2][j];
  __syncthreads();
  for (int t = 0; t < 3; ++t) {                  // output proj + residual
    float a = bo[j];
    #pragma unroll 8
    for (int d = 0; d < 64; ++d) a = fmaf(om[t][d], wo[d * 64 + j], a);
    xm[t][j] = tok[t][j] + a;
  }
  __syncthreads();
  for (int t = 0; t < 3; ++t) {                  // LN1
    float v = xm[t][j];
    float m  = block_sum64(v, red) * (1.0f / 64.0f);
    float vr = block_sum64((v - m) * (v - m), red) * (1.0f / 64.0f);
    xm[t][j] = (v - m) * rsqrtf(vr + 1e-5f) * l1g[j] + l1b[j];
    __syncthreads();
  }
  for (int t = 0; t < 3; ++t) {                  // FF up (64 -> 128)
    float s0 = fb1[j], s1 = fb1[j + 64];
    #pragma unroll 8
    for (int d = 0; d < 64; ++d) {
      float x = xm[t][d];
      s0 = fmaf(x, fw1[d * 128 + j], s0);
      s1 = fmaf(x, fw1[d * 128 + j + 64], s1);
    }
    f1[t][j] = gelu_tanh(s0); f1[t][j + 64] = gelu_tanh(s1);
  }
  __syncthreads();
  float t2v[3];
  for (int t = 0; t < 3; ++t) {                  // FF down + residual
    float s = fb2[j];
    #pragma unroll 8
    for (int d = 0; d < 128; ++d) s = fmaf(f1[t][d], fw2[d * 64 + j], s);
    t2v[t] = xm[t][j] + s;
  }
  __syncthreads();
  for (int t = 0; t < 3; ++t) {                  // LN2
    float v = t2v[t];
    float m  = block_sum64(v, red) * (1.0f / 64.0f);
    float vr = block_sum64((v - m) * (v - m), red) * (1.0f / 64.0f);
    xm[t][j] = (v - m) * rsqrtf(vr + 1e-5f) * l2g[j] + l2b[j];
    __syncthreads();
  }
  hf[j] = (xm[0][j] + xm[1][j] + xm[2][j]) * (1.0f / 3.0f);
  __syncthreads();
  if (j < 32) {                                  // head layer 1
    float s = hb1[j];
    #pragma unroll 8
    for (int d = 0; d < 64; ++d) s = fmaf(hf[d], hw1[d * 32 + j], s);
    h1s[j] = nan2num(gelu_tanh(s));
  }
  __syncthreads();
  if (j == 0) {
    float s = hb2[0];
    #pragma unroll 8
    for (int c = 0; c < 32; ++c) s = fmaf(h1s[c], hw2[c], s);
    out[b] = s;
  }
}

// -------------------------------------------------------------------- host ---

extern "C" void kernel_launch(void* const* d_in, const int* in_sizes, int n_in,
                              void* d_out, int out_size, void* d_ws, size_t ws_size,
                              hipStream_t stream) {
  (void)in_sizes; (void)n_in; (void)out_size; (void)ws_size;
  using namespace cfg;
  const float* x_temp   = (const float*)d_in[0];
  const int*   node_idx = (const int*)  d_in[1];
  const float* coords   = (const float*)d_in[2];
  const int*   edge_idx = (const int*)  d_in[3];
  const float* extras   = (const float*)d_in[4];
  const float* enc_w = (const float*)d_in[5];
  const float* enc_b = (const float*)d_in[6];
  const float* log_dt = (const float*)d_in[7];
  const float* A_re = (const float*)d_in[8];
  const float* A_im = (const float*)d_in[9];
  const float* C_re = (const float*)d_in[10];
  const float* C_im = (const float*)d_in[11];
  const float* s4D  = (const float*)d_in[12];
  const float* mix_w = (const float*)d_in[13];
  const float* mix_b = (const float*)d_in[14];
  const float* s4lg = (const float*)d_in[15];
  const float* s4lb = (const float*)d_in[16];
  const float* dec_w = (const float*)d_in[17];
  const float* dec_b = (const float*)d_in[18];
  const float* tcn_w = (const float*)d_in[19];
  const float* tcn_b = (const float*)d_in[20];
  const float* tlng = (const float*)d_in[21];
  const float* tlnb = (const float*)d_in[22];
  const float* gin_w = (const float*)d_in[23];
  const float* gin_b = (const float*)d_in[24];
  const float* gat_wl = (const float*)d_in[25];
  const float* gat_wr = (const float*)d_in[26];
  const float* gat_att = (const float*)d_in[27];
  const float* gat_bias = (const float*)d_in[28];
  const float* gat_lng = (const float*)d_in[29];
  const float* gat_lnb = (const float*)d_in[30];
  const float* ex_w1 = (const float*)d_in[31];
  const float* ex_b1 = (const float*)d_in[32];
  const float* ex_w2 = (const float*)d_in[33];
  const float* ex_b2 = (const float*)d_in[34];
  const float* wq = (const float*)d_in[35]; const float* bq = (const float*)d_in[36];
  const float* wk = (const float*)d_in[37]; const float* bk = (const float*)d_in[38];
  const float* wv = (const float*)d_in[39]; const float* bv = (const float*)d_in[40];
  const float* wo = (const float*)d_in[41]; const float* bo = (const float*)d_in[42];
  const float* l1g = (const float*)d_in[43]; const float* l1b = (const float*)d_in[44];
  const float* fw1 = (const float*)d_in[45]; const float* fb1 = (const float*)d_in[46];
  const float* fw2 = (const float*)d_in[47]; const float* fb2 = (const float*)d_in[48];
  const float* l2g = (const float*)d_in[49]; const float* l2b = (const float*)d_in[50];
  const float* hw1 = (const float*)d_in[51]; const float* hb1 = (const float*)d_in[52];
  const float* hw2 = (const float*)d_in[53]; const float* hb2 = (const float*)d_in[54];

  // workspace layout (floats) — total ≈ 59 MB
  float* W = (float*)d_ws;
  const long long HN  = (long long)HID * NST;          // 4096
  const long long BH  = (long long)BATCH * HID;        // 8192
  const long long NNH = (long long)NNODE * HID;        // 3.2M
  float* a_re  = W;            float* a_im  = a_re + HN;
  float* cf_re = a_im + HN;    float* cf_im = cf_re + HN;
  float* g_re  = cf_im + HN;   float* g_im  = g_re + HN;
  float* y_last = g_im + HN;
  float* ht = y_last + BH;
  float* hg = ht + BH;
  float* he = hg + BH;
  float* xg  = he + BH;
  float* xl  = xg + NNH;
  float* xr  = xl + NNH;
  float* agg = xr + NNH;
  float* esc = agg + NNH;
  float* emax  = esc + (long long)ETOT * 2;
  float* denom = emax + (long long)NNODE * 2;

  // temporal branch
  s4_precompute<<<(HID * NST + 255) / 256, 256, 0, stream>>>(
      log_dt, A_re, A_im, C_re, C_im, a_re, a_im, cf_re, cf_im, g_re, g_im);
  s4_scan<<<dim3(BATCH, HID), 64, 0, stream>>>(
      x_temp, a_re, a_im, cf_re, cf_im, g_re, g_im, enc_w, enc_b, s4D, y_last);
  s4_tail<<<BATCH, 64, 0, stream>>>(
      x_temp, y_last, enc_w, enc_b, mix_w, mix_b, s4lg, s4lb,
      dec_w, dec_b, tcn_w, tcn_b, tlng, tlnb, ht);

  // spatial branch
  gat_init_xg<<<(int)((NNH + 255) / 256), 256, 0, stream>>>(coords, gin_w, gin_b, xg);
  for (int layer = 0; layer < NLAY; ++layer) {
    const float* wl  = gat_wl  + (long long)layer * HID * HID;
    const float* wr  = gat_wr  + (long long)layer * HID * HID;
    const float* att = gat_att + (long long)layer * NHEAD * DHEAD;
    const float* bia = gat_bias + (long long)layer * HID;
    const float* lg  = gat_lng + (long long)layer * HID;
    const float* lb  = gat_lnb + (long long)layer * HID;
    gat_proj_wmma<<<NNODE / 16, 32, 0, stream>>>(xg, wl, wr, xl, xr);
    gat_buf_init<<<(int)((NNH + 255) / 256), 256, 0, stream>>>(agg, emax, denom);
    gat_edge_score<<<(ETOT + 7) / 8, 256, 0, stream>>>(edge_idx, xl, xr, att, esc, emax);
    gat_edge_soft<<<(ETOT * 2 + 255) / 256, 256, 0, stream>>>(edge_idx, esc, emax, denom);
    gat_edge_aggr<<<(ETOT + 7) / 8, 256, 0, stream>>>(edge_idx, xl, esc, denom, agg);
    gat_node_update<<<(NNODE + 7) / 8, 256, 0, stream>>>(agg, bia, lg, lb, xg);
  }
  gat_gather<<<(BATCH * 64 + 255) / 256, 256, 0, stream>>>(xg, node_idx, hg);

  // extras + mixer + head
  extras_mlp<<<BATCH, 64, 0, stream>>>(extras, ex_w1, ex_b1, ex_w2, ex_b2, he);
  mixer_head<<<BATCH, 64, 0, stream>>>(ht, hg, he, wq, bq, wk, bk, wv, bv, wo, bo,
                                       l1g, l1b, fw1, fb1, fw2, fb2, l2g, l2b,
                                       hw1, hb1, hw2, hb2, (float*)d_out);
}
